// HybridRBFMLP_1915555414708
// MI455X (gfx1250) — matrix operand order repaired and checked by reading, not compiled, not run on hardware
//
#include <hip/hip_runtime.h>
#include <hip/hip_bf16.h>

typedef __attribute__((ext_vector_type(16))) __bf16 v16bf;
typedef __attribute__((ext_vector_type(8)))  float  v8f;

namespace {
constexpr int Bn  = 16384;
constexpr int Dn  = 256;
constexpr int Rn  = 1024;
constexpr int H1n = 2048;
constexpr int H2n = 2048;

constexpr int MT = 128;        // workgroup M tile
constexpr int NT = 64;         // workgroup N tile
constexpr int KT = 64;         // K per stage (2 bf16 WMMAs deep)
constexpr int LDA_S = KT + 8;  // 72 bf16 = 144B stride (16B aligned, bank pad)
constexpr int LDB_S = KT + 8;
} // namespace

// ---------------------------------------------------------------- helpers
__device__ __forceinline__ unsigned lds_off_of(const void* p) {
    // LDS flat aperture keeps the LDS byte offset in addr[31:0]
    return (unsigned)(unsigned long long)p;
}

// gfx1250 async global->LDS copy, 16B per lane, tracked by ASYNCcnt
__device__ __forceinline__ void async_copy16(unsigned lds_byte, const void* sbase,
                                             unsigned gbyte) {
    asm volatile("global_load_async_to_lds_b128 %0, %1, %2"
                 :: "v"(lds_byte), "v"(gbyte), "s"(sbase)
                 : "memory");
}

// ---------------------------------------------------------------- converts
__global__ __launch_bounds__(256)
void cvt_bf16_kernel(const float* __restrict__ in, __bf16* __restrict__ out, int n) {
    int i = blockIdx.x * 256 + threadIdx.x;
    if (i < n) out[i] = (__bf16)in[i];
}

// in: [rows, cols] f32 row-major  ->  out: [cols, rows] bf16 row-major
__global__ __launch_bounds__(256)
void cvtT_bf16_kernel(const float* __restrict__ in, __bf16* __restrict__ out,
                      int rows, int cols) {
    int i = blockIdx.x * 256 + threadIdx.x;
    if (i < rows * cols) {
        int r = i / cols, c = i % cols;
        out[(size_t)c * rows + r] = (__bf16)in[i];
    }
}

// one wave per row: out[row] = sum(in[row,:]^2)
__global__ __launch_bounds__(256)
void rowsumsq_kernel(const float* __restrict__ in, float* __restrict__ out, int cols) {
    int wave = threadIdx.x >> 5, lane = threadIdx.x & 31;
    int row  = blockIdx.x * 8 + wave;
    const float* p = in + (size_t)row * cols;
    float s = 0.f;
    for (int c = lane; c < cols; c += 32) { float v = p[c]; s += v * v; }
    #pragma unroll
    for (int off = 16; off > 0; off >>= 1) s += __shfl_down(s, off, 32);
    if (lane == 0) out[row] = s;
}

// ---------------------------------------------------------------- WMMA GEMM
// C[M,N] = epilogue(A[M,K] * Bt[N,K]^T)
// MODE 0: C = exp(-(rowAux[m] + colAux[n] - 2*acc) * 0.5/sig[n]^2)   (RBF)
// MODE 1: C = relu(acc + colAux[n])                                   (Linear+ReLU)
template <int MODE>
__global__ __launch_bounds__(256)
void gemm_bf16_kernel(const __bf16* __restrict__ A,
                      const __bf16* __restrict__ Bt,
                      __bf16* __restrict__ C,
                      const float* __restrict__ rowAux,
                      const float* __restrict__ colAux,
                      const float* __restrict__ sig,
                      int M, int N, int K) {
    __shared__ __bf16 As[2][MT * LDA_S];   // 2 x 18 KB
    __shared__ __bf16 Bs[2][NT * LDB_S];   // 2 x  9 KB

    const int t    = threadIdx.x;
    const int wave = t >> 5;
    const int lane = t & 31;
    const int half = lane >> 4;   // 0 | 1
    const int l16  = lane & 15;

    const int m0 = blockIdx.x * MT;
    const int n0 = blockIdx.y * NT;
    // 8 waves arranged 4 (M) x 2 (N); each wave owns 32x32 (2x2 WMMA tiles)
    const int wm = (wave >> 1) * 32;
    const int wn = (wave & 1) * 32;

    const unsigned asBase[2] = { lds_off_of(&As[0][0]), lds_off_of(&As[1][0]) };
    const unsigned bsBase[2] = { lds_off_of(&Bs[0][0]), lds_off_of(&Bs[1][0]) };

    // per-thread async chunk coordinates (8 chunks of 16B per 64-bf16 row)
    // A tile: 128 rows -> 1024 chunks -> 4 per thread; B tile: 64 rows -> 2/thread
    auto issue_stage = [&](int buf, int k0) {
        #pragma unroll
        for (int q = 0; q < 4; ++q) {
            int c = t + 256 * q;             // 0..1023
            int row = c >> 3, col8 = (c & 7) * 8;
            unsigned g = (unsigned)(((size_t)(m0 + row) * K + k0 + col8) * 2);
            async_copy16(asBase[buf] + (unsigned)(row * LDA_S + col8) * 2, A, g);
        }
        #pragma unroll
        for (int q = 0; q < 2; ++q) {
            int c = t + 256 * q;             // 0..511
            int row = c >> 3, col8 = (c & 7) * 8;
            unsigned g = (unsigned)(((size_t)(n0 + row) * K + k0 + col8) * 2);
            async_copy16(bsBase[buf] + (unsigned)(row * LDB_S + col8) * 2, Bt, g);
        }
    };

    const v8f vzero = {0.f, 0.f, 0.f, 0.f, 0.f, 0.f, 0.f, 0.f};
    v8f acc[2][2];
    acc[0][0] = vzero; acc[0][1] = vzero; acc[1][0] = vzero; acc[1][1] = vzero;

    issue_stage(0, 0);   // prologue fill of buffer 0

    for (int k0 = 0; k0 < K; k0 += KT) {
        const int buf = (k0 / KT) & 1;
        if (k0 + KT < K) {
            issue_stage(buf ^ 1, k0 + KT);            // stream next stage
            asm volatile("s_wait_asynccnt 0x6" ::: "memory");  // current landed
        } else {
            asm volatile("s_wait_asynccnt 0x0" ::: "memory");
        }
        __syncthreads();   // all waves' async deposits visible

        // L2 warm-up for stage after next (gfx1250 global_prefetch_b8)
        if (k0 + 2 * KT < K)
            __builtin_prefetch(A + (size_t)(m0 + (t >> 1)) * K + k0 + 2 * KT, 0, 1);

        // ---- fragment loads (contiguous k = 16*half + s, same order in A and B)
        v16bf afrag[2][2], bfrag[2][2];   // [tile][kk]
        #pragma unroll
        for (int i = 0; i < 2; ++i) {
            #pragma unroll
            for (int kk = 0; kk < 2; ++kk) {
                // A 16x32: lane holds row m=l16; K halves split by lane/16
                const __bf16* ap =
                    &As[buf][(wm + i * 16 + l16) * LDA_S + kk * 32 + half * 16];
                #pragma unroll
                for (int v = 0; v < 8; ++v) {
                    int koff = v * 2;
                    afrag[i][kk][2 * v]     = ap[koff];
                    afrag[i][kk][2 * v + 1] = ap[koff + 1];
                }
                // B 32x16: lane holds col n=l16; k = half*16 + 2v + e
                const __bf16* bp =
                    &Bs[buf][(wn + i * 16 + l16) * LDB_S + kk * 32 + half * 16];
                #pragma unroll
                for (int v = 0; v < 8; ++v) {
                    bfrag[i][kk][2 * v]     = bp[2 * v];
                    bfrag[i][kk][2 * v + 1] = bp[2 * v + 1];
                }
            }
        }

        #pragma unroll
        for (int kk = 0; kk < 2; ++kk)
            #pragma unroll
            for (int i = 0; i < 2; ++i)
                #pragma unroll
                for (int j = 0; j < 2; ++j)
                    acc[i][j] = __builtin_amdgcn_wmma_f32_16x16x32_bf16(
                        false, afrag[i][kk], false, bfrag[j][kk],
                        (short)0, acc[i][j], false, false);

        __syncthreads();   // done reading buf before next iter's async overwrites
    }

    // ---- epilogue: C/D layout m = v + 8*half, n = l16
    #pragma unroll
    for (int i = 0; i < 2; ++i) {
        #pragma unroll
        for (int j = 0; j < 2; ++j) {
            const int n    = n0 + wn + j * 16 + l16;
            const float ca = colAux[n];
            float sc = 0.f;
            if (MODE == 0) { float s = sig[n]; sc = 0.5f / (s * s); }
            #pragma unroll
            for (int v = 0; v < 8; ++v) {
                const int m = m0 + wm + i * 16 + half * 8 + v;
                float a = acc[i][j][v];
                float val;
                if (MODE == 0) {
                    float d = rowAux[m] + ca - 2.0f * a;
                    val = __expf(-d * sc);
                } else {
                    val = a + ca;
                    val = val > 0.f ? val : 0.f;
                }
                C[(size_t)m * N + n] = (__bf16)val;
            }
        }
    }
}

// ---------------------------------------------------------------- final N=1 GEMV
__global__ __launch_bounds__(256)
void final_dot_kernel(const __bf16* __restrict__ H, const float* __restrict__ W3,
                      const float* __restrict__ b3, float* __restrict__ out, int K) {
    int wave = threadIdx.x >> 5, lane = threadIdx.x & 31;
    int row  = blockIdx.x * 8 + wave;
    const __bf16* p = H + (size_t)row * K;
    float s = 0.f;
    for (int c = lane; c < K; c += 32) s += (float)p[c] * W3[c];
    #pragma unroll
    for (int off = 16; off > 0; off >>= 1) s += __shfl_down(s, off, 32);
    if (lane == 0) out[row] = s + b3[0];
}

// ---------------------------------------------------------------- launcher
extern "C" void kernel_launch(void* const* d_in, const int* in_sizes, int n_in,
                              void* d_out, int out_size, void* d_ws, size_t ws_size,
                              hipStream_t stream) {
    const float* x       = (const float*)d_in[0];
    const float* centers = (const float*)d_in[1];
    const float* sigmas  = (const float*)d_in[2];
    const float* W1      = (const float*)d_in[3];
    const float* b1      = (const float*)d_in[4];
    const float* W2      = (const float*)d_in[5];
    const float* b2      = (const float*)d_in[6];
    const float* W3      = (const float*)d_in[7];
    const float* b3      = (const float*)d_in[8];
    float* out = (float*)d_out;

    char* ws = (char*)d_ws;
    size_t off = 0;
    auto carve = [&](size_t bytes) -> void* {
        void* p = ws + off;
        off += (bytes + 255) & ~(size_t)255;
        return p;
    };
    __bf16* xb  = (__bf16*)carve((size_t)Bn * Dn * 2);    //  8 MB  x bf16
    __bf16* cb  = (__bf16*)carve((size_t)Rn * Dn * 2);    //  0.5MB centers bf16
    __bf16* w1t = (__bf16*)carve((size_t)H1n * Rn * 2);   //  4 MB  W1^T bf16
    __bf16* w2t = (__bf16*)carve((size_t)H2n * H1n * 2);  //  8 MB  W2^T bf16
    float*  x2  = (float*)carve((size_t)Bn * 4);
    float*  c2  = (float*)carve((size_t)Rn * 4);
    __bf16* rbf = (__bf16*)carve((size_t)Bn * Rn * 2);    // 32 MB
    __bf16* h1  = (__bf16*)carve((size_t)Bn * H1n * 2);   // 64 MB
    __bf16* h2  = (__bf16*)carve((size_t)Bn * H2n * 2);   // 64 MB

    // pre-pass: bf16 converts (+ weight transposes) and squared norms
    cvt_bf16_kernel<<<(Bn * Dn + 255) / 256, 256, 0, stream>>>(x, xb, Bn * Dn);
    cvt_bf16_kernel<<<(Rn * Dn + 255) / 256, 256, 0, stream>>>(centers, cb, Rn * Dn);
    cvtT_bf16_kernel<<<(Rn * H1n + 255) / 256, 256, 0, stream>>>(W1, w1t, Rn, H1n);
    cvtT_bf16_kernel<<<(H1n * H2n + 255) / 256, 256, 0, stream>>>(W2, w2t, H1n, H2n);
    rowsumsq_kernel<<<Bn / 8, 256, 0, stream>>>(x, x2, Dn);
    rowsumsq_kernel<<<Rn / 8, 256, 0, stream>>>(centers, c2, Dn);

    // GEMM1 + RBF epilogue: [16384,256] x [256,1024]
    gemm_bf16_kernel<0><<<dim3(Bn / MT, Rn / NT), 256, 0, stream>>>(
        xb, cb, rbf, x2, c2, sigmas, Bn, Rn, Dn);
    // GEMM2 + bias/ReLU: [16384,1024] x [1024,2048]
    gemm_bf16_kernel<1><<<dim3(Bn / MT, H1n / NT), 256, 0, stream>>>(
        rbf, w1t, h1, nullptr, b1, nullptr, Bn, H1n, Rn);
    // GEMM3 + bias/ReLU: [16384,2048] x [2048,2048]
    gemm_bf16_kernel<1><<<dim3(Bn / MT, H2n / NT), 256, 0, stream>>>(
        h1, w2t, h2, nullptr, b2, nullptr, Bn, H2n, H1n);
    // final GEMV N=1
    final_dot_kernel<<<Bn / 8, 256, 0, stream>>>(h2, W3, b3, out, H2n);
}
